// Classifier_50208167690314
// MI455X (gfx1250) — matrix operand order repaired and checked by reading, not compiled, not run on hardware
//
#include <hip/hip_runtime.h>

typedef __attribute__((ext_vector_type(16))) __bf16 v16bf;
typedef __attribute__((ext_vector_type(8)))  float  v8f;

// ---------- helpers ----------
static __device__ __forceinline__ unsigned short f2bf(float f) {
    unsigned int x = __float_as_uint(f);
    unsigned int r = x + 0x7fffu + ((x >> 16) & 1u);   // round-to-nearest-even
    return (unsigned short)(r >> 16);
}

union Frag {
    unsigned int u[8];
    uint4        q[2];
    v16bf        v;
};

// ---------- elementwise kernels ----------
__global__ void k_f32_to_bf16(const float* __restrict__ in,
                              unsigned short* __restrict__ out, long long n) {
    long long i = (long long)blockIdx.x * blockDim.x + threadIdx.x;
    if (i < n) out[i] = f2bf(in[i]);
}

__global__ void k_zero_f32(float* __restrict__ p, long long n) {
    long long i = (long long)blockIdx.x * blockDim.x + threadIdx.x;
    if (i < n) p[i] = 0.0f;
}

// agg[dst[e]] += hW[src[e]]  (float4 per thread, 4 global f32 atomics)
__global__ void k_scatter(const float* __restrict__ hW, const int* __restrict__ src,
                          const int* __restrict__ dst, float* __restrict__ agg,
                          long long nE, int F) {
    long long idx = (long long)blockIdx.x * blockDim.x + threadIdx.x;
    int vec = F >> 2;
    long long total = nE * (long long)vec;
    if (idx >= total) return;
    long long e = idx / vec;
    int c = (int)(idx % vec) * 4;
    const float4 v = *(const float4*)(hW + (size_t)src[e] * F + c);
    float* d = agg + (size_t)dst[e] * F + c;
    atomicAdd(d + 0, v.x);
    atomicAdd(d + 1, v.y);
    atomicAdd(d + 2, v.z);
    atomicAdd(d + 3, v.w);
}

// out_bf16 = bf16(relu(agg + bias[f]))
__global__ void k_bias_relu_bf16(const float* __restrict__ agg, const float* __restrict__ bias,
                                 unsigned short* __restrict__ out, long long total, int F) {
    long long i = (long long)blockIdx.x * blockDim.x + threadIdx.x;
    if (i >= total) return;
    float v = agg[i] + bias[(int)(i % F)];
    out[i] = f2bf(v > 0.0f ? v : 0.0f);
}

// pool[n2g[node]][f] += relu(agg[node][f] + bias[f])
__global__ void k_pool(const float* __restrict__ agg, const float* __restrict__ bias,
                       const int* __restrict__ n2g, float* __restrict__ pool,
                       long long total, int F) {
    long long i = (long long)blockIdx.x * blockDim.x + threadIdx.x;
    if (i >= total) return;
    long long node = i / F;
    int f = (int)(i % F);
    float v = agg[i] + bias[f];
    if (v < 0.0f) v = 0.0f;
    atomicAdd(&pool[(size_t)n2g[node] * F + f], v);
}

__global__ void k_count(const int* __restrict__ n2g, float* __restrict__ cnt, int n) {
    int i = blockIdx.x * blockDim.x + threadIdx.x;
    if (i < n) atomicAdd(&cnt[n2g[i]], 1.0f);
}

// x[b] = bf16([pool[b]/max(cnt[b],1), descriptors[b]])
__global__ void k_concat(const float* __restrict__ pool, const float* __restrict__ cnt,
                         const float* __restrict__ desc, unsigned short* __restrict__ x,
                         int B, int F, int ND) {
    int W = F + ND;
    long long i = (long long)blockIdx.x * blockDim.x + threadIdx.x;
    if (i >= (long long)B * W) return;
    int b = (int)(i / W);
    int c = (int)(i % W);
    float v;
    if (c < F) {
        float cc = cnt[b];
        v = pool[(size_t)b * F + c] / (cc > 1.0f ? cc : 1.0f);
    } else {
        v = desc[(size_t)b * ND + (c - F)];
    }
    x[i] = f2bf(v);
}

// ---------- WMMA bf16 GEMM ----------
// C[M,N](f32 accum) = A[M,K]bf16 @ B[K,N]bf16 (+bias, +relu); out f32 and/or bf16.
// Block tile 128x64, 128 threads = 4 waves; wave w owns rows [w*32, w*32+32):
// 2 A fragments x 4 N tiles = 8 v_wmma_f32_16x16x32_bf16 per K=32 step.
// LDS layouts are fragment-ordered so every fragment is loaded with ds_load_b128:
//   sA[m][kp]        (kp = k/2; lane's A frag = 2 contiguous uint4 groups)
//   sB[n][kp] pad 20 (lane's B frag = 2 contiguous uint4 groups)
#define BM 128
#define BN 64
#define BK 32
#define BSTR 20   // padded words per B column (16 used + 4 pad; keeps 16B align, spreads banks)

__global__ __launch_bounds__(128) void k_gemm_bf16(
    const unsigned short* __restrict__ A, const unsigned short* __restrict__ B,
    const float* __restrict__ bias, float* __restrict__ Cf,
    unsigned short* __restrict__ Cbf, int M, int N, int K, int doRelu) {
    __shared__ unsigned int sA[BM * (BK / 2)];   // 128*16 u32 = 8 KB
    __shared__ unsigned int sB[BN * BSTR];       // 64*20  u32 = 5 KB

    const int tid = threadIdx.x;
    const int wave = tid >> 5;
    const int lane = tid & 31;
    const int mrow = lane & 15;   // A row in tile / B-C column in tile
    const int hi = lane >> 4;     // lane-half
    const long long bm0 = (long long)blockIdx.x * BM;
    const long long bn0 = (long long)blockIdx.y * BN;

    v8f acc[2][4] = {};

    for (int k0 = 0; k0 < K; k0 += BK) {
        // ---- stage A tile (coalesced u32 loads of bf16 pairs; K is always even) ----
        for (int i = tid; i < BM * 16; i += 128) {
            int m = i >> 4, kp = i & 15;
            long long gm = bm0 + m;
            int gk = k0 + kp * 2;
            unsigned int w = 0;
            if (gm < M && gk < K) {
                w = *(const unsigned int*)(A + (size_t)gm * K + gk);
                if (gk + BK < K)   // gfx1250 global_prefetch of next K tile
                    __builtin_prefetch(A + (size_t)gm * K + gk + BK, 0, 1);
            }
            sA[i] = w;
        }
        // ---- stage B tile fragment-ordered: sB[n*BSTR+kp] = {bf16(2kp+1,n), bf16(2kp,n)} ----
        for (int i = tid; i < 16 * (BN / 2); i += 128) {
            int kp = i >> 5;
            int n2 = (i & 31) * 2;
            int gk = k0 + kp * 2;
            long long gn = bn0 + n2;
            unsigned int lo = 0, hw = 0;
            if (gk < K) {
                if (gn + 1 < N)      lo = *(const unsigned int*)(B + (size_t)gk * N + gn);
                else if (gn < N)     lo = B[(size_t)gk * N + gn];
                if (gk + 1 < K) {
                    if (gn + 1 < N)  hw = *(const unsigned int*)(B + (size_t)(gk + 1) * N + gn);
                    else if (gn < N) hw = B[(size_t)(gk + 1) * N + gn];
                }
            }
            sB[n2 * BSTR + kp]       = (lo & 0xffffu) | (hw << 16);
            sB[(n2 + 1) * BSTR + kp] = (lo >> 16) | (hw & 0xffff0000u);
        }
        __syncthreads();

        // ---- A fragments (16x32 bf16): frag word p = sA[am*16 + hi*4 + (p>>2)*8 + (p&3)]
        //      -> two contiguous uint4 groups at +0 and +8 words.
        Frag fa[2];
#pragma unroll
        for (int mi = 0; mi < 2; ++mi) {
            const int am = wave * 32 + mi * 16 + mrow;
            const unsigned int* pA = &sA[am * 16 + hi * 4];
            fa[mi].q[0] = *(const uint4*)(pA);
            fa[mi].q[1] = *(const uint4*)(pA + 8);
        }

#pragma unroll
        for (int t = 0; t < 4; ++t) {
            // ---- B fragment (32x16 bf16): frag word p = sB[bn*BSTR + hi*4 + (p>>2)*8 + (p&3)]
            Frag fb;
            const int bn = t * 16 + mrow;
            const unsigned int* pB = &sB[bn * BSTR + hi * 4];
            fb.q[0] = *(const uint4*)(pB);
            fb.q[1] = *(const uint4*)(pB + 8);
#pragma unroll
            for (int mi = 0; mi < 2; ++mi) {
                acc[mi][t] = __builtin_amdgcn_wmma_f32_16x16x32_bf16(
                    false, fa[mi].v, false, fb.v, (short)0, acc[mi][t], false, false);
            }
        }
        __syncthreads();
    }

    // ---- epilogue: C VGPR r -> row = wave*32 + mi*16 + hi*8 + r, col = t*16 + mrow ----
#pragma unroll
    for (int mi = 0; mi < 2; ++mi) {
#pragma unroll
        for (int t = 0; t < 4; ++t) {
            long long gn = bn0 + t * 16 + mrow;
            if (gn >= N) continue;
            float bb = bias ? bias[gn] : 0.0f;
#pragma unroll
            for (int r = 0; r < 8; ++r) {
                long long gm = bm0 + wave * 32 + mi * 16 + hi * 8 + r;
                if (gm >= M) continue;
                float v = acc[mi][t][r] + bb;
                if (doRelu && v < 0.0f) v = 0.0f;
                if (Cf)  Cf[(size_t)gm * N + gn] = v;
                if (Cbf) Cbf[(size_t)gm * N + gn] = f2bf(v);
            }
        }
    }
}

// ---------- orchestration ----------
static inline long long cdivll(long long a, long long b) { return (a + b - 1) / b; }

extern "C" void kernel_launch(void* const* d_in, const int* in_sizes, int n_in,
                              void* d_out, int out_size, void* d_ws, size_t ws_size,
                              hipStream_t stream) {
    const float* features = (const float*)d_in[0];
    const float* descript = (const float*)d_in[1];
    const int*   src      = (const int*)d_in[2];
    const int*   dst      = (const int*)d_in[3];
    const int*   n2g      = (const int*)d_in[4];
    const float* W1 = (const float*)d_in[5];
    const float* b1 = (const float*)d_in[6];
    const float* W2 = (const float*)d_in[7];
    const float* b2 = (const float*)d_in[8];
    const float* lw1 = (const float*)d_in[9];
    const float* lb1 = (const float*)d_in[10];
    const float* lw2 = (const float*)d_in[11];
    const float* lb2 = (const float*)d_in[12];
    const float* cw  = (const float*)d_in[13];
    const float* cb  = (const float*)d_in[14];
    float* out = (float*)d_out;

    const int N = 100000, E = 400000, Bg = 4000;
    const int IN = 128, H = 512, ND = 200, H1 = 500, H2 = 100, NC = 2;

    // workspace carve-up (256B aligned)
    char* ws = (char*)d_ws;
    size_t off = 0;
    auto carve = [&](size_t bytes) -> void* {
        void* p = ws + off;
        off = (off + bytes + 255) & ~(size_t)255;
        return p;
    };
    unsigned short* featbf = (unsigned short*)carve((size_t)N * IN * 2);
    unsigned short* w1bf   = (unsigned short*)carve((size_t)IN * H * 2);
    unsigned short* w2bf   = (unsigned short*)carve((size_t)H * H * 2);
    unsigned short* lw1bf  = (unsigned short*)carve((size_t)(H + ND) * H1 * 2);
    unsigned short* lw2bf  = (unsigned short*)carve((size_t)H1 * H2 * 2);
    unsigned short* cwbf   = (unsigned short*)carve((size_t)H2 * NC * 2);
    float*          hW     = (float*)carve((size_t)N * H * 4);
    float*          agg    = (float*)carve((size_t)N * H * 4);
    unsigned short* hbf    = (unsigned short*)carve((size_t)N * H * 2);
    float*          pool   = (float*)carve((size_t)Bg * H * 4);
    float*          cnt    = (float*)carve((size_t)Bg * 4);
    unsigned short* xbf    = (unsigned short*)carve((size_t)Bg * (H + ND) * 2);
    unsigned short* y1bf   = (unsigned short*)carve((size_t)Bg * H1 * 2);
    unsigned short* y2bf   = (unsigned short*)carve((size_t)Bg * H2 * 2);

    const int T = 256;
    auto cvt = [&](const float* p, unsigned short* q, long long n) {
        k_f32_to_bf16<<<dim3((unsigned)cdivll(n, T)), dim3(T), 0, stream>>>(p, q, n);
    };

    // 1) bf16 conversions of inputs/weights
    cvt(features, featbf, (long long)N * IN);
    cvt(W1, w1bf, (long long)IN * H);
    cvt(W2, w2bf, (long long)H * H);
    cvt(lw1, lw1bf, (long long)(H + ND) * H1);
    cvt(lw2, lw2bf, (long long)H1 * H2);
    cvt(cw, cwbf, (long long)H2 * NC);

    auto gemm = [&](const unsigned short* A, const unsigned short* B, const float* bias,
                    float* Cf, unsigned short* Cbf, int M, int Nc, int K, int relu) {
        dim3 grid((unsigned)cdivll(M, BM), (unsigned)cdivll(Nc, BN));
        k_gemm_bf16<<<grid, dim3(128), 0, stream>>>(A, B, bias, Cf, Cbf, M, Nc, K, relu);
    };

    // 2) GCN layer 1: hW = feat @ W1 ; agg = scatter-sum ; h1 = relu(agg+b1) -> bf16
    gemm(featbf, w1bf, nullptr, hW, nullptr, N, H, IN, 0);
    long long nAgg = (long long)N * H;
    k_zero_f32<<<dim3((unsigned)cdivll(nAgg, T)), dim3(T), 0, stream>>>(agg, nAgg);
    long long nScat = (long long)E * (H / 4);
    k_scatter<<<dim3((unsigned)cdivll(nScat, T)), dim3(T), 0, stream>>>(hW, src, dst, agg, E, H);
    k_bias_relu_bf16<<<dim3((unsigned)cdivll(nAgg, T)), dim3(T), 0, stream>>>(agg, b1, hbf, nAgg, H);

    // 3) GCN layer 2
    gemm(hbf, w2bf, nullptr, hW, nullptr, N, H, H, 0);
    k_zero_f32<<<dim3((unsigned)cdivll(nAgg, T)), dim3(T), 0, stream>>>(agg, nAgg);
    k_scatter<<<dim3((unsigned)cdivll(nScat, T)), dim3(T), 0, stream>>>(hW, src, dst, agg, E, H);

    // 4) per-graph mean pooling of relu(agg+b2)
    long long nPool = (long long)Bg * H;
    k_zero_f32<<<dim3((unsigned)cdivll(nPool, T)), dim3(T), 0, stream>>>(pool, nPool);
    k_zero_f32<<<dim3((unsigned)cdivll(Bg, T)), dim3(T), 0, stream>>>(cnt, Bg);
    k_count<<<dim3((unsigned)cdivll(N, T)), dim3(T), 0, stream>>>(n2g, cnt, N);
    k_pool<<<dim3((unsigned)cdivll(nAgg, T)), dim3(T), 0, stream>>>(agg, b2, n2g, pool, nAgg, H);

    // 5) concat [mean, descriptors] -> bf16 x [Bg, 712]
    long long nX = (long long)Bg * (H + ND);
    k_concat<<<dim3((unsigned)cdivll(nX, T)), dim3(T), 0, stream>>>(pool, cnt, descript, xbf, Bg, H, ND);

    // 6) MLP head (fused bias+relu in GEMM epilogue)
    gemm(xbf,  lw1bf, lb1, nullptr, y1bf, Bg, H1, H + ND, 1);
    gemm(y1bf, lw2bf, lb2, nullptr, y2bf, Bg, H2, H1, 1);
    gemm(y2bf, cwbf,  cb,  out,     nullptr, Bg, NC, H2, 0);

    (void)in_sizes; (void)n_in; (void)out_size; (void)ws_size;
}
